// HyperbolicDistanceHead_13950053778042
// MI455X (gfx1250) — matrix operand order repaired and checked by reading, not compiled, not run on hardware
//
#include <hip/hip_runtime.h>

typedef __attribute__((ext_vector_type(16))) __bf16 v16bf;
typedef __attribute__((ext_vector_type(8)))  float v8f;

#define D_DIM 512
#define BM 128
#define BN 128
#define BK 32
#define LSTR 17  // u32 per LDS row: BK/2 data + 1 pad

__device__ __forceinline__ unsigned short f2bf(float f) {
  unsigned u = __float_as_uint(f);
  u += 0x7FFFu + ((u >> 16) & 1u);          // round-to-nearest-even
  return (unsigned short)(u >> 16);
}
__device__ __forceinline__ float bf2f(unsigned short h) {
  return __uint_as_float(((unsigned)h) << 16);
}

union FragAB { v16bf v; unsigned u[8]; };

// ---- Kernel 1: raw squared row norms of X (nx rows) and W (nw rows) ----
__global__ void row_norms_kernel(const float* __restrict__ X,
                                 const float* __restrict__ W,
                                 float* __restrict__ norms, int nx, int nw) {
  int gid  = blockIdx.x * blockDim.x + threadIdx.x;
  int wid  = gid >> 5;
  int lane = gid & 31;
  if (wid >= nx + nw) return;
  const float* src = (wid < nx) ? (X + (size_t)wid * D_DIM)
                                : (W + (size_t)(wid - nx) * D_DIM);
  const float4* s4 = (const float4*)src;
  float acc = 0.f;
#pragma unroll
  for (int i = 0; i < 4; ++i) {
    float4 v = s4[lane + i * 32];
    acc += v.x * v.x + v.y * v.y + v.z * v.z + v.w * v.w;
  }
#pragma unroll
  for (int off = 16; off > 0; off >>= 1) acc += __shfl_xor(acc, off, 32);
  if (lane == 0) norms[wid] = acc;
}

// ---- Kernel 2: split-bf16 WMMA GEMM + hyperbolic-distance epilogue ----
__global__ __launch_bounds__(256) void hyp_head_kernel(
    const float* __restrict__ X, const float* __restrict__ W,
    const float* __restrict__ xnorm, const float* __restrict__ wnorm,
    float* __restrict__ out, int V) {
  __shared__ unsigned Ahi[BM * LSTR];
  __shared__ unsigned Alo[BM * LSTR];
  __shared__ unsigned Bhi[BN * LSTR];
  __shared__ unsigned Blo[BN * LSTR];

  const int t     = threadIdx.x;
  const int lane  = t & 31;
  const int w     = t >> 5;
  const int wrow  = w >> 2;                 // 0..1 : 64-row slab
  const int wcol  = w & 3;                  // 0..3 : 32-col slab
  const int rowBase = blockIdx.y * BM;
  const int colBase = blockIdx.x * BN;
  const int mlane = lane & 15;
  const int akb   = (lane < 16) ? 0 : 4;    // A frag u32 base (K half)
  const int bkb   = (lane < 16) ? 0 : 4;    // B frag u32 base (K half)

  const v8f vzero = {0.f, 0.f, 0.f, 0.f, 0.f, 0.f, 0.f, 0.f};
  v8f c[4][2];
#pragma unroll
  for (int i = 0; i < 4; ++i)
#pragma unroll
    for (int j = 0; j < 2; ++j) c[i][j] = vzero;

  const int ldRow = t >> 3;                 // 0..31
  const int ldCol = t & 7;                  // float4 column

  for (int kk = 0; kk < D_DIM; kk += BK) {
    // Stage fp32 chunk -> LDS as bf16 hi/lo
#pragma unroll
    for (int p = 0; p < 4; ++p) {
      int r = p * 32 + ldRow;
      float4 va = ((const float4*)(X + (size_t)(rowBase + r) * D_DIM + kk))[ldCol];
      float4 vb = ((const float4*)(W + (size_t)(colBase + r) * D_DIM + kk))[ldCol];
      int idx = r * LSTR + ldCol * 2;
      unsigned short h0 = f2bf(va.x), h1 = f2bf(va.y), h2 = f2bf(va.z), h3 = f2bf(va.w);
      Ahi[idx]     = (unsigned)h0 | ((unsigned)h1 << 16);
      Ahi[idx + 1] = (unsigned)h2 | ((unsigned)h3 << 16);
      unsigned short l0 = f2bf(va.x - bf2f(h0)), l1 = f2bf(va.y - bf2f(h1));
      unsigned short l2 = f2bf(va.z - bf2f(h2)), l3 = f2bf(va.w - bf2f(h3));
      Alo[idx]     = (unsigned)l0 | ((unsigned)l1 << 16);
      Alo[idx + 1] = (unsigned)l2 | ((unsigned)l3 << 16);
      h0 = f2bf(vb.x); h1 = f2bf(vb.y); h2 = f2bf(vb.z); h3 = f2bf(vb.w);
      Bhi[idx]     = (unsigned)h0 | ((unsigned)h1 << 16);
      Bhi[idx + 1] = (unsigned)h2 | ((unsigned)h3 << 16);
      l0 = f2bf(vb.x - bf2f(h0)); l1 = f2bf(vb.y - bf2f(h1));
      l2 = f2bf(vb.z - bf2f(h2)); l3 = f2bf(vb.w - bf2f(h3));
      Blo[idx]     = (unsigned)l0 | ((unsigned)l1 << 16);
      Blo[idx + 1] = (unsigned)l2 | ((unsigned)l3 << 16);
    }
    __syncthreads();

    FragAB ahi[4], alo[4], bhi[2], blo[2];
#pragma unroll
    for (int im = 0; im < 4; ++im) {
      int base = (wrow * 64 + im * 16 + mlane) * LSTR;
#pragma unroll
      for (int v = 0; v < 4; ++v) {
        ahi[im].u[v]     = Ahi[base + akb + v];
        alo[im].u[v]     = Alo[base + akb + v];
        ahi[im].u[4 + v] = Ahi[base + akb + 8 + v];
        alo[im].u[4 + v] = Alo[base + akb + 8 + v];
      }
    }
#pragma unroll
    for (int jn = 0; jn < 2; ++jn) {
      int base = (wcol * 32 + jn * 16 + mlane) * LSTR;
#pragma unroll
      for (int v = 0; v < 8; ++v) {
        bhi[jn].u[v] = Bhi[base + bkb + v + (v & 4)];
        blo[jn].u[v] = Blo[base + bkb + v + (v & 4)];
      }
    }
#pragma unroll
    for (int im = 0; im < 4; ++im) {
#pragma unroll
      for (int jn = 0; jn < 2; ++jn) {
        c[im][jn] = __builtin_amdgcn_wmma_f32_16x16x32_bf16(
            false, ahi[im].v, false, bhi[jn].v, (short)0, c[im][jn], false, false);
        c[im][jn] = __builtin_amdgcn_wmma_f32_16x16x32_bf16(
            false, ahi[im].v, false, blo[jn].v, (short)0, c[im][jn], false, false);
        c[im][jn] = __builtin_amdgcn_wmma_f32_16x16x32_bf16(
            false, alo[im].v, false, bhi[jn].v, (short)0, c[im][jn], false, false);
      }
    }
    __syncthreads();
  }

  // ---- Epilogue: apply Poincare projection scales + hyperbolic distance ----
  const float eps   = 1e-5f;
  const int   mhalf = (lane < 16) ? 0 : 8;  // C layout: M = r + 8*(lane>=16)
#pragma unroll
  for (int im = 0; im < 4; ++im) {
    float sx[8], xns[8];
    int rows[8];
#pragma unroll
    for (int r = 0; r < 8; ++r) {
      int row = rowBase + wrow * 64 + im * 16 + mhalf + r;
      rows[r] = row;
      float n2 = xnorm[row];
      float s  = fminf((1.0f - eps) / (sqrtf(n2) + 1e-10f), 1.0f);
      sx[r]  = s;
      xns[r] = s * s * n2;
    }
#pragma unroll
    for (int jn = 0; jn < 2; ++jn) {
      int col = colBase + wcol * 32 + jn * 16 + mlane;  // C layout: N = lane%16
      float n2y = wnorm[col];
      float sy  = fminf((1.0f - eps) / (sqrtf(n2y) + 1e-10f), 1.0f);
      float yns = sy * sy * n2y;
      float omy = 1.0f - yns;
#pragma unroll
      for (int r = 0; r < 8; ++r) {
        float dot   = c[im][jn][r] * (sx[r] * sy);
        float sq    = xns[r] + yns - 2.0f * dot;
        float dist  = sqrtf(fmaxf(sq, 0.0f));
        float denom = (1.0f - xns[r]) * omy + eps;
        float arg   = 1.0f + 2.0f * dist / denom;
        arg = fmaxf(arg, 1.0f + eps);
        float ach = logf(arg + sqrtf(arg * arg - 1.0f));   // arccosh
        out[(size_t)rows[r] * V + col] = -ach;
      }
    }
  }
}

extern "C" void kernel_launch(void* const* d_in, const int* in_sizes, int n_in,
                              void* d_out, int out_size, void* d_ws, size_t ws_size,
                              hipStream_t stream) {
  (void)n_in; (void)out_size; (void)ws_size;
  const float* X = (const float*)d_in[0];   // (B*S, 512) = (4096, 512)
  const float* W = (const float*)d_in[1];   // (32000, 512)
  float* out   = (float*)d_out;             // (4096, 32000)
  float* norms = (float*)d_ws;              // [0,N): x norms, [N,N+V): w norms

  const int N = in_sizes[0] / D_DIM;        // 4096
  const int V = in_sizes[1] / D_DIM;        // 32000

  int totalRows = N + V;
  int nblk = (totalRows * 32 + 255) / 256;  // one wave32 per row
  row_norms_kernel<<<nblk, 256, 0, stream>>>(X, W, norms, N, V);

  dim3 grid(V / BN, N / BM);                // (250, 32)
  hyp_head_kernel<<<grid, 256, 0, stream>>>(X, W, norms, norms + N, out, V);
}
